// ImgtoClass_64F_31937376813207
// MI455X (gfx1250) — matrix operand order repaired and checked by reading, not compiled, not run on hardware
//
#include <hip/hip_runtime.h>
#include <hip/hip_bf16.h>

typedef __attribute__((ext_vector_type(16))) _Float16 v16h;
typedef __attribute__((ext_vector_type(8)))  _Float16 v8h;
typedef __attribute__((ext_vector_type(8)))  float    v8f;
typedef _Float16 h16;

#define BN_EPS 1e-5f

// ---------------------------------------------------------------- WMMA helpers
static __device__ __forceinline__ v8f wmma32f16(v16h a, v16h b, v8f c) {
  // D = A(16x32 f16) * B(32x16 f16) + C(16x16 f32)
  return __builtin_amdgcn_wmma_f32_16x16x32_f16(false, a, false, b, (short)0, c,
                                                false, false);
}

static __device__ __forceinline__ v16h cat8(v8h a, v8h b) {
  return __builtin_shufflevector(a, b, 0, 1, 2, 3, 4, 5, 6, 7,
                                 8, 9, 10, 11, 12, 13, 14, 15);
}

// branchless insert into sorted top-3 (t0 >= t1 >= t2): 5 VALU ops
static __device__ __forceinline__ void ins3(float x, float& t0, float& t1, float& t2) {
  float mn0 = fminf(t0, x);
  t0 = fmaxf(t0, x);
  float mn1 = fminf(t1, mn0);
  t1 = fmaxf(t1, mn0);
  t2 = fmaxf(t2, mn1);
}

// ------------------------------------------------- input f32 -> f16 NHWC, concat 164 imgs
__global__ void k_cvt(const float* __restrict__ q, const float* __restrict__ s,
                      const float* __restrict__ u, h16* __restrict__ x0) {
  const int PER = 7056;           // 84*84
  const int TOT = 164 * 3 * PER;
  int idx = blockIdx.x * 256 + threadIdx.x;  // source NCHW order (coalesced reads)
  if (idx >= TOT) return;
  int img = idx / (3 * PER);
  int r   = idx - img * 3 * PER;
  int ci  = r / PER;
  int p   = r - ci * PER;
  float v;
  if (img < 75)       v = q[idx];
  else if (img < 100) v = s[idx - 75 * 3 * PER];
  else                v = u[idx - 100 * 3 * PER];
  x0[((size_t)img * PER + p) * 3 + ci] = (h16)v;   // NHWC
}

// --------------------- weights -> n-major f16 matrix wp[cout][Kpad], k = tap*Cin + ci
__global__ void k_prep_w(const float* __restrict__ w, h16* __restrict__ wp,
                         int Cin, int Kpad) {
  int idx = blockIdx.x * 256 + threadIdx.x;
  if (idx >= Kpad * 64) return;
  int cout = idx / Kpad, k = idx - cout * Kpad;
  float v = 0.f;
  if (k < Cin * 9) {
    int tap = k / Cin, ci = k - tap * Cin;
    v = w[(cout * Cin + ci) * 9 + tap];
  }
  wp[idx] = (h16)v;
}

// ------------------------------------------------- implicit-GEMM conv3x3 SAME via WMMA
// NHWC activations; block = 256 (8 waves); wave = 16 output pixels x 64 out channels
template <int H, int W, int CIN, int KPAD>
__global__ void __launch_bounds__(256)
k_conv_wmma(const h16* __restrict__ x, const h16* __restrict__ wp,
            h16* __restrict__ y) {
  constexpr int HW = H * W;
  constexpr int KR = CIN * 9;
  __shared__ __align__(16) h16 lBt[64 * 32];   // weights, n-major [cout][kslice]
  const int img  = blockIdx.y;
  const int wave = threadIdx.x >> 5;
  const int lane = threadIdx.x & 31;
  const int lo   = lane & 15, hi = lane >> 4;
  const int pixBase = (blockIdx.x * 8 + wave) * 16;
  const h16* xi = x + (size_t)img * HW * CIN;

  v8f z = {};
  v8f acc[4];
#pragma unroll
  for (int t = 0; t < 4; t++) acc[t] = z;

  const int apix = pixBase + lo;       // A-matrix row (output pixel) for this lane
  const bool pvalid = apix < HW;
  const int py = apix / W;             // W is compile-time: mul-shift
  const int px = apix - py * W;
  const bool yT = py > 0, yB = py < H - 1;
  const bool xL = px > 0, xR = px < W - 1;
  v16h z16 = {};

  for (int kb = 0; kb < KPAD; kb += 32) {
    __syncthreads();
    {                                   // stage 64x32 weight slab, 16B per thread
      int n = threadIdx.x >> 2, ch = threadIdx.x & 3;
      *(v8h*)(lBt + n * 32 + ch * 8) =
          *(const v8h*)(wp + (size_t)n * KPAD + kb + ch * 8);
    }
    __syncthreads();

    // A fragment: 16-bit A 16x32 layout = two contiguous 8-half chunks per lane
    v16h a;
    if (CIN == 64) {
      const int tap = kb >> 6;          // tap-major K: constant tap per k-step
      const int t3  = tap / 3;
      const int dy = t3 - 1, dx = (tap - t3 * 3) - 1;
      const bool okY = (dy < 0) ? yT : ((dy > 0) ? yB : true);
      const bool okX = (dx < 0) ? xL : ((dx > 0) ? xR : true);
      const bool ok  = pvalid && okY && okX;
      if (ok) {
        const int offp = apix + dy * W + dx;          // neighbor pixel index
        const v8h* ap = (const v8h*)(xi + (size_t)offp * 64);
        const int c0 = ((kb & 63) >> 3) + hi;         // chunk of 8 channels
        a = cat8(ap[c0], ap[c0 + 2]);
      } else {
        a = z16;
      }
    } else {
#pragma unroll
      for (int v = 0; v < 8; v++) {
        const int kk = ((v < 4) ? 0 : 16) + hi * 8 + ((v & 3) << 1);
#pragma unroll
        for (int e = 0; e < 2; e++) {
          const int k = kb + kk + e;
          float val = 0.f;
          if (pvalid && k < KR) {
            const int tap = k / CIN;
            const int ci  = k - tap * CIN;
            const int t3  = tap / 3;
            const int yy = py + t3 - 1;
            const int xx = px + (tap - t3 * 3) - 1;
            if (yy >= 0 && yy < H && xx >= 0 && xx < W)
              val = (float)xi[(yy * W + xx) * CIN + ci];
          }
          a[2 * v + e] = (h16)val;
        }
      }
    }

#pragma unroll
    for (int nt = 0; nt < 4; nt++) {    // 4 x 16 out channels
      const h16* bp = lBt + (nt * 16 + lo) * 32 + hi * 8;
      v16h b = cat8(*(const v8h*)bp, *(const v8h*)(bp + 16));  // 2x ds_load_b128
      acc[nt] = wmma32f16(a, b, acc[nt]);
    }
  }

  h16* yo = y + (size_t)img * HW * 64;  // NHWC epilogue: lanes hit consecutive couts
#pragma unroll
  for (int nt = 0; nt < 4; nt++) {
    int cout = nt * 16 + lo;            // D: lane = N (channel), vgpr = M (pixel)
#pragma unroll
    for (int v = 0; v < 8; v++) {
      int pix = pixBase + v + 8 * hi;
      if (pix < HW) yo[(size_t)pix * 64 + cout] = (h16)acc[nt][v];
    }
  }
}

// ------------------------------------------------- BN batch stats: one block per (channel, group)
__global__ void __launch_bounds__(256)
k_stats(const h16* __restrict__ y, float* __restrict__ st, int HW) {
  __shared__ float ssum[256], ssq[256];
  int c = blockIdx.x, g = blockIdx.y;
  int gbase = (g == 0) ? 0 : (g == 1 ? 75 : 100);
  int gcnt  = (g == 0) ? 75 : (g == 1 ? 25 : 64);
  float s = 0.f, q = 0.f;
  for (int il = 0; il < gcnt; il++) {
    const h16* base = y + (size_t)(gbase + il) * HW * 64 + c;   // NHWC
    for (int p = threadIdx.x; p < HW; p += 256) {
      float v = (float)base[(size_t)p * 64];
      s += v; q += v * v;
    }
  }
  ssum[threadIdx.x] = s; ssq[threadIdx.x] = q;
  __syncthreads();
  for (int off = 128; off > 0; off >>= 1) {
    if (threadIdx.x < (unsigned)off) {
      ssum[threadIdx.x] += ssum[threadIdx.x + off];
      ssq[threadIdx.x]  += ssq[threadIdx.x + off];
    }
    __syncthreads();
  }
  if (threadIdx.x == 0) {
    st[(g * 64 + c) * 2]     = ssum[0];
    st[(g * 64 + c) * 2 + 1] = ssq[0];
  }
}

// --------------------- BN + LeakyReLU (+ optional 2x2 maxpool), channel-fastest (NHWC)
template <int H, int W, int POOL>
__global__ void k_bnact(const h16* __restrict__ y, h16* __restrict__ xo,
                        const float* __restrict__ st,
                        const float* __restrict__ gamma, const float* __restrict__ beta) {
  constexpr int HW  = H * W;
  constexpr int H2  = POOL ? H / 2 : H;
  constexpr int W2  = POOL ? W / 2 : W;
  constexpr int PER = H2 * W2;
  long long idx = (long long)blockIdx.x * 256 + threadIdx.x;
  if (idx >= 164LL * 64 * PER) return;
  int c   = (int)(idx & 63);
  long long r = idx >> 6;
  int p   = (int)(r % PER);
  int img = (int)(r / PER);
  int g = img < 75 ? 0 : (img < 100 ? 1 : 2);
  float cnt = (float)(g == 0 ? 75 : (g == 1 ? 25 : 64)) * (float)HW;
  float sum = st[(g * 64 + c) * 2], sq = st[(g * 64 + c) * 2 + 1];
  float mean = sum / cnt;
  float var  = sq / cnt - mean * mean;
  float sc = gamma[c] * rsqrtf(var + BN_EPS);
  float sh = beta[c] - mean * sc;
  const h16* yi = y + (size_t)img * HW * 64;
  float rr;
  if (POOL) {
    int oy = p / W2, ox = p - oy * W2;
    rr = -3.0e38f;
#pragma unroll
    for (int dy = 0; dy < 2; dy++)
#pragma unroll
      for (int dx = 0; dx < 2; dx++) {
        float v = (float)yi[(size_t)((2 * oy + dy) * W + 2 * ox + dx) * 64 + c] * sc + sh;
        v = v >= 0.f ? v : 0.2f * v;
        rr = fmaxf(rr, v);
      }
  } else {
    float v = (float)yi[(size_t)p * 64 + c] * sc + sh;
    rr = v >= 0.f ? v : 0.2f * v;
  }
  xo[((size_t)img * PER + p) * 64 + c] = (h16)rr;
}

// --------------- per-descriptor L2 normalize; NHWC in, NHWC out ([img][441][64])
__global__ void k_normdesc(const h16* __restrict__ feat, h16* __restrict__ Dn) {
  int idx = blockIdx.x * 128 + threadIdx.x;
  if (idx >= 164 * 441) return;
  const h16* f = feat + (size_t)idx * 64;
  float ss = 0.f;
  for (int c = 0; c < 64; c++) { float v = (float)f[c]; ss += v * v; }
  float inv = rsqrtf(ss);
  h16* d = Dn + (size_t)idx * 64;
  for (int c = 0; c < 64; c++) d[c] = (h16)((float)f[c] * inv);
}

// ------------------------------------------------- column maps: descriptor index img*441+pix
__global__ void k_cmap_sup(int* __restrict__ cm) {
  int idx = blockIdx.x * 256 + threadIdx.x;
  if (idx >= 5 * 2208) return;
  int j = idx / 2208, c = idx - j * 2208;
  int e = 0;
  if (c < 2205) {
    int s = c / 441, p = c - s * 441;
    e = (75 + j * 5 + s) * 441 + p;
  }
  cm[idx] = e;
}

__global__ void k_cmap_aug(const int* __restrict__ sel, int* __restrict__ cm) {
  int idx = blockIdx.x * 256 + threadIdx.x;
  if (idx >= 5 * 6624) return;
  int j = idx / 6624, c = idx - j * 6624;
  int e = 0;
  if (c < 2205) {
    int s = c / 441, p = c - s * 441;
    e = (75 + j * 5 + s) * 441 + p;
  } else if (c < 6615) {
    int r = c - 2205;
    int t = r / 441, p = r - t * 441;
    e = (100 + sel[j * 10 + t]) * 441 + p;
  }
  cm[idx] = e;
}

// ------------------------------------------------- fused WMMA GEMM + running top-3 metric
// block = (image, class); 8 waves x 4 row-tiles; per-column A gather (4x b128) feeds 8 WMMAs.
// Main loop is unmasked; the single partial tile at the end handles the mValid boundary.
__global__ void __launch_bounds__(256)
k_metric(const h16* __restrict__ Dn, const int* __restrict__ cmap, int cmPitch,
         int mValid, int rowBase, float* __restrict__ out) {
  __shared__ float psum[8];
  const int img  = blockIdx.x;
  const int cls  = blockIdx.y;
  const int wave = threadIdx.x >> 5;
  const int lane = threadIdx.x & 31;
  const int lo   = lane & 15, hi = lane >> 4;
  const int* cm = cmap + cls * cmPitch;
  const int qbase = (rowBase + img) * 441;

  v16h b0[4], b1[4];                    // 4 row-tiles of query descriptors (B), 2 k-steps
  float t0[4], t1[4], t2[4];
#pragma unroll
  for (int j = 0; j < 4; j++) {
    const int qrow = (wave + 8 * j) * 16 + lo;
    const int qp = qrow < 441 ? qrow : 440;   // clamp; invalid rows masked at the end
    const v8h* dp = (const v8h*)(Dn + (size_t)(qbase + qp) * 64);
    b0[j] = cat8(dp[hi], dp[2 + hi]);        // k = hi*8..+7, 16+hi*8..+7
    b1[j] = cat8(dp[4 + hi], dp[6 + hi]);    // k = 32+hi*8..+7, 48+hi*8..+7
    t0[j] = t1[j] = t2[j] = -1e30f;
  }

  const int fullTiles = mValid >> 4;    // tiles with all 16 columns valid
  for (int ct = 0; ct < fullTiles; ct++) {
    __builtin_prefetch(cm + ct * 16 + 16 + lo, 0, 0);   // global_prefetch_b8
    int e = cm[ct * 16 + lo];           // this lane's A row (support descriptor)
    const v8h* ap = (const v8h*)(Dn + (size_t)e * 64);
    v16h a0 = cat8(ap[hi], ap[2 + hi]);              // chunks per A swizzle
    v16h a1 = cat8(ap[4 + hi], ap[6 + hi]);
#pragma unroll
    for (int j = 0; j < 4; j++) {
      v8f c = {};
      c = wmma32f16(a0, b0[j], c);
      c = wmma32f16(a1, b1[j], c);
#pragma unroll
      for (int v = 0; v < 8; v++)       // unmasked: pure 5-op insert per value
        ins3(c[v], t0[j], t1[j], t2[j]);
    }
  }
  if (fullTiles * 16 < mValid) {        // single masked tail tile
    const int ct = fullTiles;
    int e = cm[ct * 16 + lo];
    const v8h* ap = (const v8h*)(Dn + (size_t)e * 64);
    v16h a0 = cat8(ap[hi], ap[2 + hi]);
    v16h a1 = cat8(ap[4 + hi], ap[6 + hi]);
#pragma unroll
    for (int j = 0; j < 4; j++) {
      v8f c = {};
      c = wmma32f16(a0, b0[j], c);
      c = wmma32f16(a1, b1[j], c);
#pragma unroll
      for (int v = 0; v < 8; v++) {     // D: lane n = query row, M = v + 8*hi
        int mcol = ct * 16 + v + 8 * hi;
        float x = (mcol < mValid) ? c[v] : -1e30f;   // branchless mask + insert
        ins3(x, t0[j], t1[j], t2[j]);
      }
    }
  }

  float waveAcc = 0.f;
#pragma unroll
  for (int j = 0; j < 4; j++) {
    // merge the two half-lanes covering the same query row
    float o0 = __shfl_xor(t0[j], 16);
    float o1 = __shfl_xor(t1[j], 16);
    float o2 = __shfl_xor(t2[j], 16);
    ins3(o0, t0[j], t1[j], t2[j]);
    ins3(o1, t0[j], t1[j], t2[j]);
    ins3(o2, t0[j], t1[j], t2[j]);
    const int qrow = (wave + 8 * j) * 16 + lo;
    float contrib = (hi == 0 && qrow < 441) ? (t0[j] + t1[j] + t2[j]) : 0.f;
#pragma unroll
    for (int off = 16; off > 0; off >>= 1) contrib += __shfl_xor(contrib, off);
    waveAcc += contrib;
  }
  if (lane == 0) psum[wave] = waveAcc;
  __syncthreads();
  if (threadIdx.x == 0) {
    float s = 0.f;
    for (int i = 0; i < 8; i++) s += psum[i];
    out[img * 5 + cls] = s;
  }
}

// ------------------------------------------------- softmax over 5 classes
__global__ void k_softmax5(const float* __restrict__ simu, float* __restrict__ probs) {
  int i = threadIdx.x;
  if (i >= 64) return;
  float v[5]; float mx = -1e30f;
  for (int j = 0; j < 5; j++) { v[j] = simu[i * 5 + j]; mx = fmaxf(mx, v[j]); }
  float s = 0.f;
  for (int j = 0; j < 5; j++) { v[j] = expf(v[j] - mx); s += v[j]; }
  for (int j = 0; j < 5; j++) probs[i * 5 + j] = v[j] / s;
}

// ------------------------------------------------- top-10 unlabeled per class
__global__ void k_top10(const float* __restrict__ probs, int* __restrict__ sel) {
  int j = threadIdx.x;
  if (j >= 5) return;
  float vals[64];
  for (int i = 0; i < 64; i++) vals[i] = probs[i * 5 + j];
  for (int t = 0; t < 10; t++) {
    int bi = 0; float bv = vals[0];
    for (int i = 1; i < 64; i++)
      if (vals[i] > bv) { bv = vals[i]; bi = i; }
    sel[j * 10 + t] = bi;
    vals[bi] = -1e30f;
  }
}

// ================================================================ host launch
extern "C" void kernel_launch(void* const* d_in, const int* in_sizes, int n_in,
                              void* d_out, int out_size, void* d_ws, size_t ws_size,
                              hipStream_t stream) {
  (void)in_sizes; (void)n_in; (void)out_size; (void)ws_size;
  const float* in1 = (const float*)d_in[0];   // [75,3,84,84]
  const float* in2 = (const float*)d_in[1];   // [5,5,3,84,84]
  const float* in3 = (const float*)d_in[2];   // [64,3,84,84]
  const float* w[4] = {(const float*)d_in[3], (const float*)d_in[4],
                       (const float*)d_in[5], (const float*)d_in[6]};
  const float* g[4] = {(const float*)d_in[7], (const float*)d_in[9],
                       (const float*)d_in[11], (const float*)d_in[13]};
  const float* bb[4] = {(const float*)d_in[8], (const float*)d_in[10],
                        (const float*)d_in[12], (const float*)d_in[14]};
  float* out = (float*)d_out;                 // [75,5]
  char* ws = (char*)d_ws;

  // workspace layout (bytes); Y1 reused for Y2
  constexpr size_t OFF_X0   = 0;                         // 6.95 MB
  constexpr size_t OFF_Y1   = 8ull << 20;                // 148.1 MB
  constexpr size_t OFF_Y2   = 8ull << 20;                // alias of Y1 (Y1 dead)
  constexpr size_t OFF_X2   = 48ull << 20;               // 9.26 MB
  constexpr size_t OFF_Y3   = 60ull << 20;
  constexpr size_t OFF_X3   = 70ull << 20;
  constexpr size_t OFF_Y4   = 80ull << 20;
  constexpr size_t OFF_FEAT = 90ull << 20;
  constexpr size_t OFF_DN   = 100ull << 20;
  constexpr size_t OFF_WP1  = 110ull << 20;
  constexpr size_t OFF_WP2  = OFF_WP1 + (128ull << 10);
  constexpr size_t OFF_WP3  = OFF_WP2 + (128ull << 10);
  constexpr size_t OFF_WP4  = OFF_WP3 + (128ull << 10);
  constexpr size_t OFF_ST   = OFF_WP4 + (128ull << 10);  // 4 layers * 384 f32
  constexpr size_t OFF_SIMU = OFF_ST + (64ull << 10);
  constexpr size_t OFF_PROB = OFF_SIMU + (16ull << 10);
  constexpr size_t OFF_SEL  = OFF_PROB + (16ull << 10);
  constexpr size_t OFF_CMS  = OFF_SEL + (16ull << 10);   // 5*2208 ints
  constexpr size_t OFF_CMA  = OFF_CMS + (64ull << 10);   // 5*6624 ints
  constexpr size_t OFF_X1   = 160ull << 20;              // 37 MB (above Y1 end)

  h16* x0   = (h16*)(ws + OFF_X0);
  h16* y1   = (h16*)(ws + OFF_Y1);
  h16* x1   = (h16*)(ws + OFF_X1);
  h16* y2   = (h16*)(ws + OFF_Y2);
  h16* x2   = (h16*)(ws + OFF_X2);
  h16* y3   = (h16*)(ws + OFF_Y3);
  h16* x3   = (h16*)(ws + OFF_X3);
  h16* y4   = (h16*)(ws + OFF_Y4);
  h16* feat = (h16*)(ws + OFF_FEAT);
  h16* dn   = (h16*)(ws + OFF_DN);
  h16* wp[4] = {(h16*)(ws + OFF_WP1), (h16*)(ws + OFF_WP2),
                (h16*)(ws + OFF_WP3), (h16*)(ws + OFF_WP4)};
  float* st   = (float*)(ws + OFF_ST);
  float* simu = (float*)(ws + OFF_SIMU);
  float* prob = (float*)(ws + OFF_PROB);
  int*   sel  = (int*)(ws + OFF_SEL);
  int*   cmS  = (int*)(ws + OFF_CMS);
  int*   cmA  = (int*)(ws + OFF_CMA);

  // 1) convert inputs (NHWC), prep weights (n-major, tap-major K)
  k_cvt<<<(164 * 3 * 7056 + 255) / 256, 256, 0, stream>>>(in1, in2, in3, x0);
  k_prep_w<<<(32 * 64 + 255) / 256, 256, 0, stream>>>(w[0], wp[0], 3, 32);
  for (int l = 1; l < 4; l++)
    k_prep_w<<<(576 * 64 + 255) / 256, 256, 0, stream>>>(w[l], wp[l], 64, 576);

  // 2) feature extractor
  auto stats = [&](const h16* yin, float* stl, int HW) {
    k_stats<<<dim3(64, 3), 256, 0, stream>>>(yin, stl, HW);
  };

  k_conv_wmma<84, 84, 3, 32><<<dim3((7056 + 127) / 128, 164), 256, 0, stream>>>(x0, wp[0], y1);
  stats(y1, st + 0 * 384, 7056);
  k_bnact<84, 84, 1><<<(int)((164LL * 64 * 1764 + 255) / 256), 256, 0, stream>>>(
      y1, x1, st + 0 * 384, g[0], bb[0]);                // -> [164,1764,64]

  k_conv_wmma<42, 42, 64, 576><<<dim3((1764 + 127) / 128, 164), 256, 0, stream>>>(x1, wp[1], y2);
  stats(y2, st + 1 * 384, 1764);
  k_bnact<42, 42, 1><<<(int)((164LL * 64 * 441 + 255) / 256), 256, 0, stream>>>(
      y2, x2, st + 1 * 384, g[1], bb[1]);                // -> [164,441,64]

  k_conv_wmma<21, 21, 64, 576><<<dim3((441 + 127) / 128, 164), 256, 0, stream>>>(x2, wp[2], y3);
  stats(y3, st + 2 * 384, 441);
  k_bnact<21, 21, 0><<<(int)((164LL * 64 * 441 + 255) / 256), 256, 0, stream>>>(
      y3, x3, st + 2 * 384, g[2], bb[2]);

  k_conv_wmma<21, 21, 64, 576><<<dim3((441 + 127) / 128, 164), 256, 0, stream>>>(x3, wp[3], y4);
  stats(y4, st + 3 * 384, 441);
  k_bnact<21, 21, 0><<<(int)((164LL * 64 * 441 + 255) / 256), 256, 0, stream>>>(
      y4, feat, st + 3 * 384, g[3], bb[3]);

  // 3) normalized descriptors (descriptor-major, 128B each) for all 164 images
  k_normdesc<<<(164 * 441 + 127) / 128, 128, 0, stream>>>(feat, dn);

  // 4) semi-supervised stage: unlabeled vs support banks, top-3 metric + softmax + top-10
  k_cmap_sup<<<(5 * 2208 + 255) / 256, 256, 0, stream>>>(cmS);
  k_metric<<<dim3(64, 5), 256, 0, stream>>>(dn, cmS, 2208, 2205, 100, simu);
  k_softmax5<<<1, 64, 0, stream>>>(simu, prob);
  k_top10<<<1, 8, 0, stream>>>(prob, sel);
  k_cmap_aug<<<(5 * 6624 + 255) / 256, 256, 0, stream>>>(sel, cmA);

  // 5) final query metric vs augmented banks -> d_out [75,5]
  k_metric<<<dim3(75, 5), 256, 0, stream>>>(dn, cmA, 6624, 6615, 0, out);
}
